// SnliModel_49881750176406
// MI455X (gfx1250) — matrix operand order repaired and checked by reading, not compiled, not run on hardware
//
#include <hip/hip_runtime.h>
#include <hip/hip_bf16.h>

// ---------------- types ----------------
typedef __bf16 bf16_t;
typedef __attribute__((ext_vector_type(4)))  __bf16  v4bf;
typedef __attribute__((ext_vector_type(8)))  __bf16  v8bf;
typedef __attribute__((ext_vector_type(16))) __bf16  v16bf;
typedef __attribute__((ext_vector_type(8)))  float   v8f;

#define HH   256
#define EE   300
#define EPAD 320
#define VV   50000
#define BSZ  32
#define SS   64
#define TT   64
#define AMBN 8
#define PP   128        // S + T
#define N5H  1280       // 5*H
#define K2H  512        // 2*H

__device__ __forceinline__ float sigf(float x) { return 1.0f / (1.0f + expf(-x)); }

// ---------------------------------------------------------------------------
// Build fragment-major bf16 B-operand from row-major f32 weights.
// B tile is K=32 x N=16. Lane L: N = nt*16 + (L&15); K = kt*32 + (e<8 ? h*8+e : 16+h*8+(e-8)), h=L>>4.
// Flat layout: frag(kt*80+nt) * 512 + L*16 + e  -> each lane reads 32 contiguous B.
// ---------------------------------------------------------------------------
__global__ void build_bfrag_kernel(const float* __restrict__ src, bf16_t* __restrict__ dst,
                                   int kValid, int srcPitch, int total) {
  int g = blockIdx.x * 256 + threadIdx.x;
  if (g >= total) return;
  int f  = g >> 9;
  int rr = g & 511;
  int L  = rr >> 4;
  int e  = rr & 15;
  int kt = f / 80;
  int nt = f % 80;
  int n  = nt * 16 + (L & 15);
  int k  = kt * 32 + ((L < 16) ? 0 : 8) + ((e < 8) ? e : (8 + e));
  float v = (k < kValid) ? src[(size_t)n * srcPitch + k] : 0.0f;
  dst[g] = (bf16_t)v;
}

// ---------------------------------------------------------------------------
// Embedding gather + f32->bf16, K padded 300->320 with zeros.
// ---------------------------------------------------------------------------
__global__ void embed_kernel(const int* __restrict__ s1, const int* __restrict__ s2,
                             const float* __restrict__ glove, const float* __restrict__ unk,
                             bf16_t* __restrict__ embBf) {
  int g = blockIdx.x * 256 + threadIdx.x;
  const int total = 2 * BSZ * SS * EPAD;
  if (g >= total) return;
  int sent = g / (BSZ * SS * EPAD);
  int rem  = g % (BSZ * SS * EPAD);
  int row  = rem / EPAD;
  int k    = rem % EPAD;
  const int* ss = sent ? s2 : s1;
  int word = ss[row];
  float v = 0.0f;
  if (k < EE) {
    if (word >= 0) {
      int w = word; if (w > VV - 1) w = VV - 1;
      v = glove[(size_t)w * EE + k];
    } else {
      v = unk[k];
    }
  }
  embBf[g] = (bf16_t)v;
}

// ---------------------------------------------------------------------------
// un = energy_u / max(||energy_u||, eps)
// ---------------------------------------------------------------------------
__global__ void prep_un_kernel(const float* __restrict__ eu, float* __restrict__ un) {
  __shared__ float red[256];
  int tid = threadIdx.x;
  float v = eu[tid];
  red[tid] = v * v;
  __syncthreads();
  for (int s = 128; s > 0; s >>= 1) { if (tid < s) red[tid] += red[tid + s]; __syncthreads(); }
  un[tid] = v / fmaxf(sqrtf(red[0]), 1e-8f);
}

// ---------------------------------------------------------------------------
// Leaf kernel: preact = emb @ W^T + b ; c0=i*u ; h0=o*tanh(c0) -> chart[:, :64]
// ---------------------------------------------------------------------------
__global__ __launch_bounds__(256) void leaf_kernel(const bf16_t* __restrict__ embBf,
                                                   const bf16_t* __restrict__ wfrag,
                                                   const float*  __restrict__ bias,
                                                   float*        __restrict__ chart) {
  const int sent = blockIdx.x >> 7;
  const int rt   = blockIdx.x & 127;
  const int tid  = threadIdx.x;
  const int wave = tid >> 5;
  const int lane = tid & 31;
  const int mhalf = lane & 15;
  const bool hiL = lane >= 16;

  const bf16_t* Abase = embBf + (size_t)sent * (BSZ * SS) * EPAD + (size_t)(rt * 16) * EPAD;

  v8f acc[10];
  for (int i = 0; i < 10; i++)
    for (int q = 0; q < 8; q++) acc[i][q] = 0.0f;

  for (int kt = 0; kt < EPAD / 32; ++kt) {
    const int k1 = kt * 32 + (hiL ? 8 : 0);
    const bf16_t* ar = Abase + mhalf * EPAD + k1;
    v8bf lo = *(const v8bf*)(ar);
    v8bf hv = *(const v8bf*)(ar + 16);
    v16bf af;
#pragma unroll
    for (int i = 0; i < 8; i++) { af[i] = lo[i]; af[8 + i] = hv[i]; }
#pragma unroll
    for (int g = 0; g < 5; ++g) {
#pragma unroll
      for (int j = 0; j < 2; ++j) {
        int nt = g * 16 + wave * 2 + j;
        v16bf bf = *(const v16bf*)(wfrag + ((size_t)(kt * 80 + nt) << 9) + lane * 16);
        acc[g * 2 + j] = __builtin_amdgcn_wmma_f32_16x16x32_bf16(
            false, af, false, bf, (short)0, acc[g * 2 + j], false, false);
      }
    }
  }

#pragma unroll
  for (int j = 0; j < 2; j++) {
    const int hcol = wave * 32 + j * 16 + mhalf;
    const float bi = bias[hcol];
    const float bo = bias[768 + hcol];
    const float bu = bias[1024 + hcol];
#pragma unroll
    for (int r = 0; r < 8; r++) {
      int row  = hiL ? (8 + r) : r;
      float i_ = sigf(acc[j][r] + bi);
      float o_ = sigf(acc[6 + j][r] + bo);
      float u_ = tanhf(acc[8 + j][r] + bu);
      float c  = i_ * u_;                 // ccL = ccR = 0 at leaves
      float h  = o_ * tanhf(c);
      int grow = rt * 16 + row;
      int b = grow >> 6, s = grow & 63;
      size_t base = (((size_t)sent * BSZ + b) * PP + s) * K2H;
      chart[base + hcol]       = c;
      chart[base + HH + hcol]  = h;
    }
  }
}

// ---------------------------------------------------------------------------
// One tree-LSTM step (both sentences). grid = 32; block = 256 (8 waves);
// each block owns 2 batch items (16 M-rows). Wave w owns n-tiles
// { g*16 + w*2 + j } so all five gates for its 32 H-columns stay in registers.
// LDS: A operand (16KB, aliased by reduction buffer after GEMM) + ccL/ccR (32KB).
// ---------------------------------------------------------------------------
__global__ __launch_bounds__(256) void step_kernel(float* __restrict__ chart,
                                                   const bf16_t* __restrict__ ufrag,
                                                   const float*  __restrict__ bias,
                                                   const int*    __restrict__ ops1,
                                                   const int*    __restrict__ ops2,
                                                   const float*  __restrict__ un,
                                                   int t) {
  __shared__ __align__(32) char ldsBuf[16 * K2H * 2];   // 16 KB: A operand, then partials
  __shared__ float ldsCL[16 * HH];                      // 16 KB
  __shared__ float ldsCR[16 * HH];                      // 16 KB
  __shared__ float eRow[16];
  __shared__ float sprob[16];

  bf16_t* ldsA = (bf16_t*)ldsBuf;
  float (*partials)[16][128] = (float (*)[16][128])ldsBuf;   // aliases ldsA (post-GEMM)

  const int sent = blockIdx.x >> 4;
  const int pair = blockIdx.x & 15;
  const int b0   = pair * 2;
  const int* ops = sent ? ops2 : ops1;
  const int tid  = threadIdx.x;
  const size_t sentBase = (size_t)sent * BSZ * PP * K2H;

  // ---- gather hh (h halves) into LDS as bf16, float4-vectorized ----
  for (int g4 = tid; g4 < 2048; g4 += 256) {
    int r  = g4 >> 7;                 // 128 float4 groups per row
    int k  = (g4 & 127) * 4;
    int b  = b0 + (r >> 3);
    int a  = r & 7;
    int lr = k >> 8;
    int idx = ops[(((b * TT) + t) * AMBN + a) * 2 + lr];
    idx = idx < 0 ? 0 : (idx > PP - 1 ? PP - 1 : idx);
    const float4 v = *(const float4*)&chart[sentBase + ((size_t)b * PP + idx) * K2H + HH + (k & 255)];
    v4bf pk;
    pk[0] = (bf16_t)v.x; pk[1] = (bf16_t)v.y; pk[2] = (bf16_t)v.z; pk[3] = (bf16_t)v.w;
    *(v4bf*)&ldsA[r * K2H + k] = pk;
  }
  // ---- stage ccL / ccR (c halves) into LDS, float4-vectorized ----
  for (int g4 = tid; g4 < 2048; g4 += 256) {
    int lr  = g4 >> 10;               // 1024 groups per side
    int rem = g4 & 1023;
    int r   = rem >> 6;               // 64 float4 groups per row
    int k   = (rem & 63) * 4;
    int b   = b0 + (r >> 3);
    int a   = r & 7;
    int idx = ops[(((b * TT) + t) * AMBN + a) * 2 + lr];
    idx = idx < 0 ? 0 : (idx > PP - 1 ? PP - 1 : idx);
    const float4 v = *(const float4*)&chart[sentBase + ((size_t)b * PP + idx) * K2H + k];
    float* dst = (lr ? ldsCR : ldsCL) + r * HH + k;
    *(float4*)dst = v;
  }
  __syncthreads();

  const int wave = tid >> 5;
  const int lane = tid & 31;
  const int mhalf = lane & 15;
  const bool hiL = lane >= 16;

  v8f acc[10];
  for (int i = 0; i < 10; i++)
    for (int q = 0; q < 8; q++) acc[i][q] = 0.0f;

  for (int kt = 0; kt < K2H / 32; ++kt) {
    const int k1 = kt * 32 + (hiL ? 8 : 0);
    const bf16_t* ar = &ldsA[mhalf * K2H + k1];
    v8bf lo = *(const v8bf*)(ar);
    v8bf hv = *(const v8bf*)(ar + 16);
    v16bf af;
#pragma unroll
    for (int i = 0; i < 8; i++) { af[i] = lo[i]; af[8 + i] = hv[i]; }
#pragma unroll
    for (int g = 0; g < 5; ++g) {
#pragma unroll
      for (int j = 0; j < 2; ++j) {
        int nt = g * 16 + wave * 2 + j;
        v16bf bf = *(const v16bf*)(ufrag + ((size_t)(kt * 80 + nt) << 9) + lane * 16);
        acc[g * 2 + j] = __builtin_amdgcn_wmma_f32_16x16x32_bf16(
            false, af, false, bf, (short)0, acc[g * 2 + j], false, false);
      }
    }
  }
  __syncthreads();   // ldsA dead; its space becomes `partials`

  // ---- register-resident LSTM cell + norm/dot partials ----
  float cV[2][8], hV[2][8], psq[8], pdot[8];
#pragma unroll
  for (int r = 0; r < 8; r++) { psq[r] = 0.0f; pdot[r] = 0.0f; }

#pragma unroll
  for (int j = 0; j < 2; j++) {
    const int hcol = wave * 32 + j * 16 + mhalf;
    const float bi  = bias[hcol];
    const float bfl = bias[256 + hcol];
    const float bfr = bias[512 + hcol];
    const float bo  = bias[768 + hcol];
    const float bu  = bias[1024 + hcol];
    const float unv = un[hcol];
#pragma unroll
    for (int r = 0; r < 8; r++) {
      int row = hiL ? (8 + r) : r;
      float ccL = ldsCL[row * HH + hcol];
      float ccR = ldsCR[row * HH + hcol];
      float i_ = sigf(acc[j][r]     + bi);
      float fL = sigf(acc[2 + j][r] + bfl);
      float fR = sigf(acc[4 + j][r] + bfr);
      float o_ = sigf(acc[6 + j][r] + bo);
      float u_ = tanhf(acc[8 + j][r] + bu);
      float c = fL * ccL + fR * ccR + i_ * u_;
      float h = o_ * tanhf(c);
      cV[j][r] = c; hV[j][r] = h;
      psq[r]  += h * h;
      pdot[r] += h * unv;
    }
  }
  {
    int col = wave * 16 + mhalf;
#pragma unroll
    for (int r = 0; r < 8; r++) {
      int row = hiL ? (8 + r) : r;
      partials[0][row][col] = psq[r];
      partials[1][row][col] = pdot[r];
    }
  }
  __syncthreads();
  if (tid < 16) {
    float sq = 0.0f, dt = 0.0f;
    for (int c2 = 0; c2 < 128; c2++) { sq += partials[0][tid][c2]; dt += partials[1][tid][c2]; }
    eRow[tid] = dt / fmaxf(sqrtf(sq), 1e-8f);
  }
  __syncthreads();
  if (tid < 2) {
    float m = -1e30f;
    for (int a = 0; a < 8; a++) m = fmaxf(m, eRow[tid * 8 + a]);
    float ex[8]; float ssum = 0.0f;
    for (int a = 0; a < 8; a++) { ex[a] = expf(eRow[tid * 8 + a] - m); ssum += ex[a]; }
    for (int a = 0; a < 8; a++) sprob[tid * 8 + a] = ex[a] / ssum;
  }
  __syncthreads();
#pragma unroll
  for (int j = 0; j < 2; j++) {
    const int hcol = wave * 32 + j * 16 + mhalf;
    const int bb = hiL ? 1 : 0;
    float sc = 0.0f, sh = 0.0f;
#pragma unroll
    for (int r = 0; r < 8; r++) {
      float w = sprob[bb * 8 + r];
      sc += w * cV[j][r];
      sh += w * hV[j][r];
    }
    int b = b0 + bb;
    size_t o = sentBase + ((size_t)b * PP + (SS + t)) * K2H;
    chart[o + hcol]      = sc;
    chart[o + HH + hcol] = sh;
  }
}

// ---------------------------------------------------------------------------
// MLP layer 1: internal1 = relu(conc @ W1^T + b1)   (33 MFLOP — VALU is fine)
// ---------------------------------------------------------------------------
__global__ void final1_kernel(const float* __restrict__ chart, const float* __restrict__ W1,
                              const float* __restrict__ b1, float* __restrict__ internal1) {
  int g = blockIdx.x * 256 + threadIdx.x;
  if (g >= BSZ * 1024) return;
  int b = g >> 10, m = g & 1023;
  float sum = b1[m];
  for (int j = 0; j < 512; j++) {
    int sent = j >> 8;
    int col  = HH + (j & 255);
    float cv = chart[(((size_t)sent * BSZ + b) * PP + (PP - 1)) * K2H + col];
    sum += cv * W1[(size_t)m * 512 + j];
  }
  internal1[g] = fmaxf(sum, 0.0f);
}

// ---------------------------------------------------------------------------
// MLP layer 2 + softmax over 3 classes.
// ---------------------------------------------------------------------------
__global__ void final2_kernel(const float* __restrict__ internal1, const float* __restrict__ W2,
                              const float* __restrict__ b2, float* __restrict__ out) {
  __shared__ float red[128];
  __shared__ float logits[3];
  int b = blockIdx.x, tid = threadIdx.x;
  for (int cls = 0; cls < 3; ++cls) {
    float p = 0.0f;
    for (int m = tid; m < 1024; m += 128) p += internal1[b * 1024 + m] * W2[cls * 1024 + m];
    red[tid] = p;
    __syncthreads();
    for (int s = 64; s > 0; s >>= 1) { if (tid < s) red[tid] += red[tid + s]; __syncthreads(); }
    if (tid == 0) logits[cls] = red[0] + b2[cls];
    __syncthreads();
  }
  if (tid == 0) {
    float m = fmaxf(logits[0], fmaxf(logits[1], logits[2]));
    float e0 = expf(logits[0] - m), e1 = expf(logits[1] - m), e2 = expf(logits[2] - m);
    float s = e0 + e1 + e2;
    out[b * 3 + 0] = e0 / s;
    out[b * 3 + 1] = e1 / s;
    out[b * 3 + 2] = e2 / s;
  }
}

// ---------------------------------------------------------------------------
extern "C" void kernel_launch(void* const* d_in, const int* in_sizes, int n_in,
                              void* d_out, int out_size, void* d_ws, size_t ws_size,
                              hipStream_t stream) {
  const int*   sent1 = (const int*)d_in[0];
  const int*   ops1  = (const int*)d_in[1];
  const int*   sent2 = (const int*)d_in[3];
  const int*   ops2  = (const int*)d_in[4];
  const float* glove = (const float*)d_in[6];
  const float* W     = (const float*)d_in[7];
  const float* U     = (const float*)d_in[8];
  const float* b     = (const float*)d_in[9];
  const float* eu    = (const float*)d_in[10];
  const float* unk   = (const float*)d_in[11];
  const float* W1    = (const float*)d_in[12];
  const float* b1    = (const float*)d_in[13];
  const float* W2    = (const float*)d_in[14];
  const float* b2    = (const float*)d_in[15];
  float* out = (float*)d_out;

  // workspace layout (bytes, all 256-aligned)
  const size_t CHART_B = 2ull * BSZ * PP * K2H * 4;        // 16,777,216
  const size_t UFRAG_B = 16ull * 80 * 512 * 2;             //  1,310,720
  const size_t WFRAG_B = 10ull * 80 * 512 * 2;             //    819,200
  const size_t EMB_B   = 2ull * BSZ * SS * EPAD * 2;       //  2,621,440
  const size_t UN_B    = 256 * 4;

  char* ws = (char*)d_ws;
  float*  chart     = (float*)(ws);
  bf16_t* ufrag     = (bf16_t*)(ws + CHART_B);
  bf16_t* wfrag     = (bf16_t*)(ws + CHART_B + UFRAG_B);
  bf16_t* embBf     = (bf16_t*)(ws + CHART_B + UFRAG_B + WFRAG_B);
  float*  un        = (float*)(ws + CHART_B + UFRAG_B + WFRAG_B + EMB_B);
  float*  internal1 = (float*)(ws + CHART_B + UFRAG_B + WFRAG_B + EMB_B + UN_B);

  hipMemsetAsync(chart, 0, CHART_B, stream);

  build_bfrag_kernel<<<(16 * 80 * 512 + 255) / 256, 256, 0, stream>>>(U, ufrag, 512, 512, 16 * 80 * 512);
  build_bfrag_kernel<<<(10 * 80 * 512 + 255) / 256, 256, 0, stream>>>(W, wfrag, 300, 300, 10 * 80 * 512);
  prep_un_kernel<<<1, 256, 0, stream>>>(eu, un);
  embed_kernel<<<(2 * BSZ * SS * EPAD + 255) / 256, 256, 0, stream>>>(sent1, sent2, glove, unk, embBf);

  leaf_kernel<<<2 * 128, 256, 0, stream>>>(embBf, wfrag, b, chart);

  for (int t = 0; t < TT; ++t)
    step_kernel<<<32, 256, 0, stream>>>(chart, ufrag, b, ops1, ops2, un, t);

  final1_kernel<<<(BSZ * 1024 + 255) / 256, 256, 0, stream>>>(chart, W1, b1, internal1);
  final2_kernel<<<BSZ, 128, 0, stream>>>(internal1, W2, b2, out);
}
